// SVCNN_65970697666562
// MI455X (gfx1250) — matrix operand order repaired and checked by reading, not compiled, not run on hardware
//
#include <hip/hip_runtime.h>

#define DIMK 1024
#define NQ   4096
#define NS   65536
#define TOPK 4

#define MTILE   128           // queries per block
#define NCHUNKS 64            // grid.x
#define NCHUNK  (NS/NCHUNKS)  // 1024 synth rows per block
#define NSTEP   64            // synth rows per inner step
#define KTILE   64            // K staged per LDS round
#define KSTEPS  (DIMK/KTILE)  // 16 (even)

typedef _Float16 half4v  __attribute__((ext_vector_type(4)));
typedef _Float16 half8v  __attribute__((ext_vector_type(8)));
typedef _Float16 half16v __attribute__((ext_vector_type(16)));
typedef float    v8f     __attribute__((ext_vector_type(8)));
typedef int      v4i     __attribute__((ext_vector_type(4)));

#if defined(__has_builtin)
#if __has_builtin(__builtin_amdgcn_global_load_async_to_lds_b128)
#define HAVE_ASYNC_LDS 1
#endif
#if __has_builtin(__builtin_amdgcn_s_wait_asynccnt)
#define HAVE_WAIT_ASYNC 1
#endif
#endif

#define AS1G __attribute__((address_space(1)))
#define AS3L __attribute__((address_space(3)))

__device__ __forceinline__ void async_copy_b128(const _Float16* g, _Float16* l)
{
#if defined(HAVE_ASYNC_LDS)
    // signature (probe-derived): (int4 AS1*, int4 AS3*, imm offset, imm cpol)
    __builtin_amdgcn_global_load_async_to_lds_b128(
        (AS1G v4i*)(g), (AS3L v4i*)(l), 0, 0);
#else
    *(half8v*)l = *(const half8v*)g;   // fallback: VGPR round-trip + ds_store
#endif
}

__device__ __forceinline__ void wait_async0()
{
#if defined(HAVE_WAIT_ASYNC)
    __builtin_amdgcn_s_wait_asynccnt(0);
#else
    asm volatile("s_wait_asynccnt 0" ::: "memory");  // harmless if counter unused
#endif
}

__device__ __forceinline__ void top4_insert(float v, int idx,
    float& v0, float& v1, float& v2, float& v3,
    int& i0, int& i1, int& i2, int& i3)
{
    if (v > v3) {
        if (v > v0)      { v3=v2;i3=i2; v2=v1;i2=i1; v1=v0;i1=i0; v0=v;i0=idx; }
        else if (v > v1) { v3=v2;i3=i2; v2=v1;i2=i1; v1=v; i1=idx; }
        else if (v > v2) { v3=v2;i3=i2; v2=v; i2=idx; }
        else             { v3=v; i3=idx; }
    }
}

// ---- Kernel A (f16 path): fused f32->f16 convert (S and Q) + 1/||s|| -------
// one wave32 per row; rows [0,NS) = S, rows [NS, NS+NQ) = Q
__global__ __launch_bounds__(256)
void prep_kernel(const float* __restrict__ Q, const float* __restrict__ S,
                 float* __restrict__ innorm,
                 _Float16* __restrict__ Qh, _Float16* __restrict__ Sh)
{
    const int r    = (blockIdx.x * blockDim.x + threadIdx.x) >> 5;
    const int lane = threadIdx.x & 31;
    if (r >= NS + NQ) return;
    const bool isS = r < NS;
    const float*    src = isS ? S  + (size_t)r * DIMK : Q  + (size_t)(r - NS) * DIMK;
    _Float16*       dst = isS ? Sh + (size_t)r * DIMK : Qh + (size_t)(r - NS) * DIMK;
    float s = 0.f;
    #pragma unroll
    for (int i = 0; i < DIMK / (32 * 4); ++i) {
        const int off = (i * 32 + lane) * 4;
        float4 f = *(const float4*)(src + off);
        s += f.x*f.x + f.y*f.y + f.z*f.z + f.w*f.w;
        half4v h = { (_Float16)f.x, (_Float16)f.y, (_Float16)f.z, (_Float16)f.w };
        *(half4v*)(dst + off) = h;
    }
    #pragma unroll
    for (int off = 16; off > 0; off >>= 1) s += __shfl_xor(s, off, 32);
    if (lane == 0 && isS) innorm[r] = rsqrtf(s);
}

// ---- Kernel A (fallback path): norms only ----------------------------------
__global__ __launch_bounds__(256)
void norm_kernel(const float* __restrict__ S, float* __restrict__ innorm)
{
    const int wv   = (blockIdx.x * blockDim.x + threadIdx.x) >> 5;
    const int lane = threadIdx.x & 31;
    if (wv >= NS) return;
    const float* row = S + (size_t)wv * DIMK;
    float s = 0.f;
    #pragma unroll
    for (int i = 0; i < DIMK / (32 * 4); ++i) {
        float4 f = *(const float4*)(row + (i * 32 + lane) * 4);
        s += f.x*f.x + f.y*f.y + f.z*f.z + f.w*f.w;
    }
    #pragma unroll
    for (int off = 16; off > 0; off >>= 1) s += __shfl_xor(s, off, 32);
    if (lane == 0) innorm[wv] = rsqrtf(s);
}

// ---- Kernel B (f16 path): async-staged, double-buffered WMMA + top-4 -------
// grid = (NCHUNKS, NQ/MTILE), block = 256 (8 wave32)
__global__ __launch_bounds__(256)
void sim_topk_f16(const _Float16* __restrict__ Qh, const _Float16* __restrict__ Sh,
                  const float* __restrict__ innorm,
                  float* __restrict__ pvals, int* __restrict__ pidx)
{
    __shared__ alignas(16) _Float16 Abuf[2][MTILE * KTILE];  // 2 x 16 KB
    __shared__ alignas(16) _Float16 Bbuf[2][NSTEP * KTILE];  // 2 x  8 KB
    __shared__ float simbuf[MTILE * NSTEP];                  // 32 KB

    const int tid  = threadIdx.x;
    const int wave = tid >> 5;
    const int lane = tid & 31;
    const int hi   = lane >> 4;
    const int l15  = lane & 15;
    const int m0   = blockIdx.y * MTILE;
    const int nbase = blockIdx.x * NCHUNK;

    float v0=-1e30f, v1=-1e30f, v2=-1e30f, v3=-1e30f;
    int   i0=0, i1=0, i2=0, i3=0;

    for (int step = 0; step < NCHUNK / NSTEP; ++step) {
        const int n0 = nbase + step * NSTEP;
        v8f acc[4] = { {0,0,0,0,0,0,0,0}, {0,0,0,0,0,0,0,0},
                       {0,0,0,0,0,0,0,0}, {0,0,0,0,0,0,0,0} };

        // stage (buf, kb): A tile MTILE x KTILE, B tile NSTEP x KTILE, f16
        auto stage = [&](int buf, int kb) {
            #pragma unroll
            for (int j = 0; j < 4; ++j) {                 // 1024 b128 groups
                const int g = tid + 256 * j;
                const int row = g >> 3, c8 = (g & 7) * 8;
                async_copy_b128(Qh + (size_t)(m0 + row) * DIMK + kb * KTILE + c8,
                                &Abuf[buf][row * KTILE + c8]);
            }
            #pragma unroll
            for (int j = 0; j < 2; ++j) {                 // 512 b128 groups
                const int g = tid + 256 * j;
                const int row = g >> 3, c8 = (g & 7) * 8;
                async_copy_b128(Sh + (size_t)(n0 + row) * DIMK + kb * KTILE + c8,
                                &Bbuf[buf][row * KTILE + c8]);
            }
        };

        // one KTILE of WMMA work from a fixed LDS buffer pair
        auto do_tile = [&](const _Float16* Ab, const _Float16* Bb) {
            #pragma unroll
            for (int ks = 0; ks < KTILE / 32; ++ks) {
                // A 16x32 f16 layout: lane half 'hi' -> K = hi*8 + e (e<8),
                // K = 16 + hi*8 + (e-8) otherwise; row m = l15
                const _Float16* ap = &Ab[(wave * 16 + l15) * KTILE + ks * 32 + hi * 8];
                half8v alo = *(const half8v*)ap;
                half8v ahi = *(const half8v*)(ap + 16);
                half16v a = __builtin_shufflevector(alo, ahi,
                    0,1,2,3,4,5,6,7,8,9,10,11,12,13,14,15);
                #pragma unroll
                for (int nt = 0; nt < 4; ++nt) {
                    // B 32x16 f16 layout: column n = l15, K = hi*8 + e (e<8), 16 + hi*8 + (e-8) otherwise
                    const _Float16* bp = &Bb[(nt * 16 + l15) * KTILE + ks * 32 + hi * 8];
                    half8v blo = *(const half8v*)bp;
                    half8v bhi = *(const half8v*)(bp + 16);
                    half16v b = __builtin_shufflevector(blo, bhi,
                        0,1,2,3,4,5,6,7,8,9,10,11,12,13,14,15);
                    acc[nt] = __builtin_amdgcn_wmma_f32_16x16x32_f16(
                        false, a, false, b, (short)0, acc[nt], false, false);
                }
            }
        };

        stage(0, 0);
        #pragma unroll 1
        for (int kb = 0; kb < KSTEPS; kb += 2) {
            // ---- buffer 0 holds stage kb ----
            wait_async0();       // my stage(kb) landed in LDS
            __syncthreads();     // everyone's landed; previous compute done
            stage(1, kb + 1);    // KSTEPS even => kb+1 always valid; overlaps compute
            do_tile(Abuf[0], Bbuf[0]);
            // ---- buffer 1 holds stage kb+1 ----
            wait_async0();
            __syncthreads();
            if (kb + 2 < KSTEPS) stage(0, kb + 2);
            do_tile(Abuf[1], Bbuf[1]);
        }

        // scale by 1/||s||, dump sims: C layout lane l: M = r + hi*8, N = l15
        const float sc0 = innorm[n0 +  0 + l15];
        const float sc1 = innorm[n0 + 16 + l15];
        const float sc2 = innorm[n0 + 32 + l15];
        const float sc3 = innorm[n0 + 48 + l15];
        #pragma unroll
        for (int r = 0; r < 8; ++r) {
            const int m = wave * 16 + hi * 8 + r;
            simbuf[m * NSTEP +  0 + l15] = acc[0][r] * sc0;
            simbuf[m * NSTEP + 16 + l15] = acc[1][r] * sc1;
            simbuf[m * NSTEP + 32 + l15] = acc[2][r] * sc2;
            simbuf[m * NSTEP + 48 + l15] = acc[3][r] * sc3;
        }
        __syncthreads();
        if (tid < MTILE) {
            #pragma unroll 8
            for (int j = 0; j < NSTEP; ++j)
                top4_insert(simbuf[tid * NSTEP + j], n0 + j,
                            v0, v1, v2, v3, i0, i1, i2, i3);
        }
        __syncthreads();
    }

    if (tid < MTILE) {
        const size_t base = ((size_t)(m0 + tid) * NCHUNKS + blockIdx.x) * 4;
        pvals[base + 0] = v0; pvals[base + 1] = v1;
        pvals[base + 2] = v2; pvals[base + 3] = v3;
        pidx[base + 0] = i0; pidx[base + 1] = i1;
        pidx[base + 2] = i2; pidx[base + 3] = i3;
    }
}

// ---- Kernel B (fallback path): f32-staged WMMA (converts in LDS stage) -----
__global__ __launch_bounds__(256)
void sim_topk_f32(const float* __restrict__ Q, const float* __restrict__ S,
                  const float* __restrict__ innorm,
                  float* __restrict__ pvals, int* __restrict__ pidx)
{
    __shared__ alignas(16) _Float16 As[MTILE * KTILE];
    __shared__ alignas(16) _Float16 Bs[NSTEP * KTILE];
    __shared__ float simbuf[MTILE * NSTEP];

    const int tid  = threadIdx.x;
    const int wave = tid >> 5;
    const int lane = tid & 31;
    const int hi   = lane >> 4;
    const int l15  = lane & 15;
    const int m0   = blockIdx.y * MTILE;
    const int nbase = blockIdx.x * NCHUNK;

    float v0=-1e30f, v1=-1e30f, v2=-1e30f, v3=-1e30f;
    int   i0=0, i1=0, i2=0, i3=0;

    for (int step = 0; step < NCHUNK / NSTEP; ++step) {
        const int n0 = nbase + step * NSTEP;
        v8f acc[4] = { {0,0,0,0,0,0,0,0}, {0,0,0,0,0,0,0,0},
                       {0,0,0,0,0,0,0,0}, {0,0,0,0,0,0,0,0} };

        for (int kb = 0; kb < KSTEPS; ++kb) {
            __syncthreads();
            for (int i = tid * 4; i < MTILE * KTILE; i += 256 * 4) {
                const int row = i >> 6, col = i & (KTILE - 1);
                const float4 f = *(const float4*)(Q + (size_t)(m0 + row) * DIMK + kb * KTILE + col);
                half4v h = { (_Float16)f.x, (_Float16)f.y, (_Float16)f.z, (_Float16)f.w };
                *(half4v*)&As[row * KTILE + col] = h;
            }
            for (int i = tid * 4; i < NSTEP * KTILE; i += 256 * 4) {
                const int row = i >> 6, col = i & (KTILE - 1);
                const float4 f = *(const float4*)(S + (size_t)(n0 + row) * DIMK + kb * KTILE + col);
                half4v h = { (_Float16)f.x, (_Float16)f.y, (_Float16)f.z, (_Float16)f.w };
                *(half4v*)&Bs[row * KTILE + col] = h;
            }
            __syncthreads();
            #pragma unroll
            for (int ks = 0; ks < KTILE / 32; ++ks) {
                const _Float16* ap = &As[(wave * 16 + l15) * KTILE + ks * 32 + hi * 8];
                half8v alo = *(const half8v*)ap;
                half8v ahi = *(const half8v*)(ap + 16);
                half16v a = __builtin_shufflevector(alo, ahi,
                    0,1,2,3,4,5,6,7,8,9,10,11,12,13,14,15);
                #pragma unroll
                for (int nt = 0; nt < 4; ++nt) {
                    const _Float16* bp = &Bs[(nt * 16 + l15) * KTILE + ks * 32 + hi * 8];
                    half8v blo = *(const half8v*)bp;
                    half8v bhi = *(const half8v*)(bp + 16);
                    half16v b = __builtin_shufflevector(blo, bhi,
                        0,1,2,3,4,5,6,7,8,9,10,11,12,13,14,15);
                    acc[nt] = __builtin_amdgcn_wmma_f32_16x16x32_f16(
                        false, a, false, b, (short)0, acc[nt], false, false);
                }
            }
        }
        const float sc0 = innorm[n0 +  0 + l15];
        const float sc1 = innorm[n0 + 16 + l15];
        const float sc2 = innorm[n0 + 32 + l15];
        const float sc3 = innorm[n0 + 48 + l15];
        #pragma unroll
        for (int r = 0; r < 8; ++r) {
            const int m = wave * 16 + hi * 8 + r;
            simbuf[m * NSTEP +  0 + l15] = acc[0][r] * sc0;
            simbuf[m * NSTEP + 16 + l15] = acc[1][r] * sc1;
            simbuf[m * NSTEP + 32 + l15] = acc[2][r] * sc2;
            simbuf[m * NSTEP + 48 + l15] = acc[3][r] * sc3;
        }
        __syncthreads();
        if (tid < MTILE) {
            #pragma unroll 8
            for (int j = 0; j < NSTEP; ++j)
                top4_insert(simbuf[tid * NSTEP + j], n0 + j,
                            v0, v1, v2, v3, i0, i1, i2, i3);
        }
        __syncthreads();
    }

    if (tid < MTILE) {
        const size_t base = ((size_t)(m0 + tid) * NCHUNKS + blockIdx.x) * 4;
        pvals[base + 0] = v0; pvals[base + 1] = v1;
        pvals[base + 2] = v2; pvals[base + 3] = v3;
        pidx[base + 0] = i0; pidx[base + 1] = i1;
        pidx[base + 2] = i2; pidx[base + 3] = i3;
    }
}

// ---- Kernel C: merge partials, gather mean (f32 source), repeat x2 ---------
__global__ __launch_bounds__(256)
void merge_gather_kernel(const float* __restrict__ S,
                         const float* __restrict__ pvals,
                         const int* __restrict__ pidx,
                         float* __restrict__ out)
{
    __shared__ float lv[NCHUNKS * TOPK];
    __shared__ int   li[NCHUNKS * TOPK];
    __shared__ int   best[TOPK];

    const int q = blockIdx.x;
    const int tid = threadIdx.x;
    lv[tid] = pvals[(size_t)q * NCHUNKS * TOPK + tid];
    li[tid] = pidx[(size_t)q * NCHUNKS * TOPK + tid];
    __syncthreads();

    if (tid == 0) {
        float v0=-1e30f, v1=-1e30f, v2=-1e30f, v3=-1e30f;
        int   i0=0, i1=0, i2=0, i3=0;
        for (int j = 0; j < NCHUNKS * TOPK; ++j)
            top4_insert(lv[j], li[j], v0, v1, v2, v3, i0, i1, i2, i3);
        best[0]=i0; best[1]=i1; best[2]=i2; best[3]=i3;
    }
    __syncthreads();

    const float* r0 = S + (size_t)best[0] * DIMK;
    const float* r1 = S + (size_t)best[1] * DIMK;
    const float* r2 = S + (size_t)best[2] * DIMK;
    const float* r3 = S + (size_t)best[3] * DIMK;
    for (int col = tid; col < DIMK; col += 256) {
        const float m = 0.25f * (r0[col] + r1[col] + r2[col] + r3[col]);
        out[(size_t)(2 * q)     * DIMK + col] = m;
        out[(size_t)(2 * q + 1) * DIMK + col] = m;
    }
}

extern "C" void kernel_launch(void* const* d_in, const int* in_sizes, int n_in,
                              void* d_out, int out_size, void* d_ws, size_t ws_size,
                              hipStream_t stream)
{
    const float* Q = (const float*)d_in[0];   // (4096, 1024) f32
    const float* S = (const float*)d_in[1];   // (65536, 1024) f32
    float* out = (float*)d_out;               // (8192, 1024) f32

    char* ws = (char*)d_ws;
    const size_t off_innorm = 0;
    const size_t off_pvals  = off_innorm + (size_t)NS * 4;
    const size_t off_pidx   = off_pvals  + (size_t)NQ * NCHUNKS * TOPK * 4;
    const size_t off_qh     = off_pidx   + (size_t)NQ * NCHUNKS * TOPK * 4;
    const size_t off_sh     = off_qh     + (size_t)NQ * DIMK * 2;
    const size_t need_f16   = off_sh     + (size_t)NS * DIMK * 2;

    float*    innorm = (float*)(ws + off_innorm);
    float*    pvals  = (float*)(ws + off_pvals);
    int*      pidx   = (int*)  (ws + off_pidx);
    _Float16* Qh     = (_Float16*)(ws + off_qh);
    _Float16* Sh     = (_Float16*)(ws + off_sh);

    dim3 grid(NCHUNKS, NQ / MTILE);
    if (ws_size >= need_f16) {
        prep_kernel<<<(NS + NQ) / 8, 256, 0, stream>>>(Q, S, innorm, Qh, Sh);
        sim_topk_f16<<<grid, 256, 0, stream>>>(Qh, Sh, innorm, pvals, pidx);
    } else {
        norm_kernel<<<NS / 8, 256, 0, stream>>>(S, innorm);
        sim_topk_f32<<<grid, 256, 0, stream>>>(Q, S, innorm, pvals, pidx);
    }
    merge_gather_kernel<<<NQ, 256, 0, stream>>>(S, pvals, pidx, out);
}
